// TransAtt_10926396801707
// MI455X (gfx1250) — matrix operand order repaired and checked by reading, not compiled, not run on hardware
//
#include <hip/hip_runtime.h>
#include <math.h>

// ---------------- CDNA5 WMMA plumbing (gfx1250, wave32) ----------------
typedef _Float16 f16;
typedef _Float16 v16h __attribute__((ext_vector_type(16)));
typedef float    v8f  __attribute__((ext_vector_type(8)));

__device__ __forceinline__ v8f wmma_f16(v16h a, v16h b, v8f c) {
  // D = A(16x32 f16) * B(32x16 f16) + C(16x16 f32)
  return __builtin_amdgcn_wmma_f32_16x16x32_f16(false, a, false, b, (short)0, c, false, false);
}

// A fragment (16x32) from LDS f16 tile, row-major, stride ld.
// lanes 0-15: rows 0..15, K pairs {0..7, 16..23}; lanes 16-31: same rows, K+8.
__device__ __forceinline__ v16h lds_a(const f16* p, int ld, int lane) {
  int row  = lane & 15;
  int koff = (lane & 16) ? 8 : 0;
  v16h a = {};
#pragma unroll
  for (int j = 0; j < 8; ++j) {
    int k = ((j < 4) ? (2 * j) : (16 + 2 * (j - 4))) + koff;
    a[2 * j]     = p[row * ld + k];
    a[2 * j + 1] = p[row * ld + k + 1];
  }
  return a;
}
// B fragment (32x16) from LDS f16 tile stored [k][n], stride ld. N = lane%16, K pairs as A.
__device__ __forceinline__ v16h lds_b(const f16* p, int ld, int lane) {
  int n    = lane & 15;
  int koff = (lane & 16) ? 8 : 0;
  v16h b = {};
#pragma unroll
  for (int j = 0; j < 8; ++j) {
    int k = ((j < 4) ? (2 * j) : (16 + 2 * (j - 4))) + koff;
    b[2 * j]     = p[k * ld + n];
    b[2 * j + 1] = p[(k + 1) * ld + n];
  }
  return b;
}
// A fragment from global f32, row-major stride ld, optional scale.
__device__ __forceinline__ v16h g_a(const float* p, int ld, int lane, float s) {
  int row  = lane & 15;
  int koff = (lane & 16) ? 8 : 0;
  v16h a = {};
#pragma unroll
  for (int j = 0; j < 8; ++j) {
    int k = ((j < 4) ? (2 * j) : (16 + 2 * (j - 4))) + koff;
    a[2 * j]     = (f16)(s * p[row * ld + k]);
    a[2 * j + 1] = (f16)(s * p[row * ld + k + 1]);
  }
  return a;
}
// B fragment from global f32 stored [k][n] (row-major, stride ld): N = lane%16, K pairs as A.
__device__ __forceinline__ v16h g_b(const float* p, int ld, int lane) {
  int n    = lane & 15;
  int koff = (lane & 16) ? 8 : 0;
  v16h b = {};
#pragma unroll
  for (int j = 0; j < 8; ++j) {
    int k = ((j < 4) ? (2 * j) : (16 + 2 * (j - 4))) + koff;
    b[2 * j]     = (f16)p[k * ld + n];
    b[2 * j + 1] = (f16)p[(k + 1) * ld + n];
  }
  return b;
}
// B fragment from global f32 stored [n][k] (i.e. we need X^T): element[K][N=lane%16] = p[N*ld + K], K pairs as A.
__device__ __forceinline__ v16h g_bT(const float* p, int ld, int lane) {
  int n    = lane & 15;
  int koff = (lane & 16) ? 8 : 0;
  v16h b = {};
#pragma unroll
  for (int j = 0; j < 8; ++j) {
    int k = ((j < 4) ? (2 * j) : (16 + 2 * (j - 4))) + koff;
    b[2 * j]     = (f16)p[n * ld + k];
    b[2 * j + 1] = (f16)p[n * ld + k + 1];
  }
  return b;
}

__device__ __forceinline__ float apply_act(float x, int act) {
  if (act == 1) return fmaxf(x, 0.f);
  if (act == 2) return 0.5f * x * (1.f + erff(x * 0.70710678118654752f)); // exact GELU
  if (act == 3) return tanhf(x);
  return x;
}

// ---------------- Generic WMMA GEMM ----------------
// C = act(scale*(A@B) + diag*I + bias) + resid.  Row-major, K%32==0.
// FULL=true: M%128==0 and N%128==0 -> unguarded float4 (b128) staging & epilogue only.
// FULL=false: guarded scalar path for ragged tiles (used once per layer, N=64 GEMM).
// Block: 256 threads = 8 waves (2 along M x 4 along N); block tile 128x128;
// each wave owns 64x32 = 8 WMMA accumulators per K-step. Next K-slab is prefetched
// into L2 via global_prefetch_b8 while the current one computes.
template <bool FULL>
__global__ void gemm_wmma(const float* __restrict__ A, const float* __restrict__ B,
                          const float* __restrict__ bias, const float* __restrict__ resid,
                          float* __restrict__ C, int M, int N, int K,
                          long long sA, long long sB, long long sC,
                          float scale, float diag, int act) {
  const int lane = threadIdx.x & 31;
  const int wid  = threadIdx.x >> 5;
  const int wm   = wid & 1;   // 2 waves along M (64 rows each)
  const int wn   = wid >> 1;  // 4 waves along N (32 cols each)
  const int row0 = blockIdx.y * 128;
  const int col0 = blockIdx.x * 128;
  A += (long long)blockIdx.z * sA;
  B += (long long)blockIdx.z * sB;
  C += (long long)blockIdx.z * sC;
  const float* R = resid ? resid + (long long)blockIdx.z * sC : nullptr;

  __shared__ f16 As[128][33];
  __shared__ f16 Bs[32][136];
  v8f acc[4][2];
  {
    v8f z = {};
#pragma unroll
    for (int i = 0; i < 4; ++i) { acc[i][0] = z; acc[i][1] = z; }
  }

  for (int k0 = 0; k0 < K; k0 += 32) {
    if (FULL) {
      // ---- stage A 128x32: 1024 float4 loads (global_load_b128), no guards ----
#pragma unroll
      for (int i = 0; i < 4; ++i) {
        int t = threadIdx.x + i * 256;
        int r = t >> 3, c = (t & 7) << 2;
        float4 v4 = *(const float4*)&A[(long long)(row0 + r) * K + k0 + c];
        As[r][c]     = (f16)v4.x;
        As[r][c + 1] = (f16)v4.y;
        As[r][c + 2] = (f16)v4.z;
        As[r][c + 3] = (f16)v4.w;
      }
      // ---- stage B 32x128 ----
#pragma unroll
      for (int i = 0; i < 4; ++i) {
        int t = threadIdx.x + i * 256;
        int r = t >> 5, c = (t & 31) << 2;
        float4 v4 = *(const float4*)&B[(long long)(k0 + r) * N + col0 + c];
        Bs[r][c]     = (f16)v4.x;
        Bs[r][c + 1] = (f16)v4.y;
        Bs[r][c + 2] = (f16)v4.z;
        Bs[r][c + 3] = (f16)v4.w;
      }
      // prefetch next K-slab into L2 (global_prefetch_b8) while this one is consumed
      if (k0 + 32 < K) {
        for (int t = threadIdx.x; t < 128; t += 256)
          __builtin_prefetch(&A[(long long)(row0 + t) * K + k0 + 32], 0, 1);
        for (int t = threadIdx.x; t < 32; t += 256)
          __builtin_prefetch(&B[(long long)(k0 + 32 + t) * N + col0], 0, 1);
      }
    } else {
      for (int t = threadIdx.x; t < 128 * 32; t += 256) {
        int r = t >> 5, c = t & 31;
        int gr = row0 + r;
        As[r][c] = (gr < M) ? (f16)A[(long long)gr * K + k0 + c] : (f16)0.f;
      }
      for (int t = threadIdx.x; t < 32 * 128; t += 256) {
        int r = t >> 7, c = t & 127;
        int gc = col0 + c;
        Bs[r][c] = (gc < N) ? (f16)B[(long long)(k0 + r) * N + gc] : (f16)0.f;
      }
    }
    __syncthreads();
    v16h b0 = lds_b(&Bs[0][wn * 32], 136, lane);
    v16h b1 = lds_b(&Bs[0][wn * 32 + 16], 136, lane);
#pragma unroll
    for (int mt = 0; mt < 4; ++mt) {
      v16h a = lds_a(&As[wm * 64 + mt * 16][0], 33, lane);
      acc[mt][0] = wmma_f16(a, b0, acc[mt][0]);
      acc[mt][1] = wmma_f16(a, b1, acc[mt][1]);
    }
    __syncthreads();
  }

  const int rsub = (lane >> 4) << 3;
  const int csub = lane & 15;
#pragma unroll
  for (int mt = 0; mt < 4; ++mt) {
#pragma unroll
    for (int nt = 0; nt < 2; ++nt) {
      int colg = col0 + wn * 32 + nt * 16 + csub;
      if (!FULL && colg >= N) continue;
      float bs = bias ? bias[colg] : 0.f;
#pragma unroll
      for (int r = 0; r < 8; ++r) {
        int gr = row0 + wm * 64 + mt * 16 + rsub + r;
        if (!FULL && gr >= M) continue;
        float v = acc[mt][nt][r] * scale;
        if (diag != 0.f && gr == colg) v += diag;
        v = apply_act(v + bs, act);
        if (R) v += R[(long long)gr * N + colg];
        C[(long long)gr * N + colg] = v;
      }
    }
  }
}

// ---------------- Small glue kernels ----------------
// LayerNorm over last dim (512). One block (256 thr) per row.
__global__ void layernorm512(const float* __restrict__ x, const float* __restrict__ g,
                             const float* __restrict__ b, float* __restrict__ y) {
  int row = blockIdx.x, t = threadIdx.x;
  const float* xr = x + (long long)row * 512;
  float a0 = xr[t], a1 = xr[t + 256];
  __shared__ float red[256];
  red[t] = a0 + a1; __syncthreads();
  for (int o = 128; o > 0; o >>= 1) { if (t < o) red[t] += red[t + o]; __syncthreads(); }
  float mu = red[0] * (1.f / 512.f); __syncthreads();
  float d0 = a0 - mu, d1 = a1 - mu;
  red[t] = d0 * d0 + d1 * d1; __syncthreads();
  for (int o = 128; o > 0; o >>= 1) { if (t < o) red[t] += red[t + o]; __syncthreads(); }
  float inv = rsqrtf(red[0] * (1.f / 512.f) + 1e-5f);
  float* yr = y + (long long)row * 512;
  yr[t]       = d0 * inv * g[t] + b[t];
  yr[t + 256] = d1 * inv * g[t + 256] + b[t + 256];
}

// Landmark means: q_l (with 1/sqrt(d) scale folded in) and k_l. block=(bh*256+lm), 64 thr (d).
__global__ void landmarks_k(const float* __restrict__ qkv, float* __restrict__ ql,
                            float* __restrict__ kl) {
  int blk = blockIdx.x;
  int bh = blk >> 8, lm = blk & 255;
  int batch = bh >> 3, head = bh & 7;
  int d = threadIdx.x;
  const float* base = qkv + ((long long)batch * 16384 + lm * 64) * 1536 + head * 64 + d;
  float sq = 0.f, sk = 0.f;
  for (int j = 0; j < 64; ++j) {
    sq += base[(long long)j * 1536];
    sk += base[(long long)j * 1536 + 512];
  }
  long long o = ((long long)bh * 256 + lm) * 64 + d;
  ql[o] = sq * (0.125f / 64.f);   // q scale 64^-0.5 folded into landmark mean
  kl[o] = sk * (1.f / 64.f);
}

// attn2 = softmax(q_l @ k_l^T) rows. grid(256 rows, 16 bh), 256 thr (cols).
__global__ void attn2_softmax(const float* __restrict__ ql, const float* __restrict__ kl,
                              float* __restrict__ a2) {
  int row = blockIdx.x, bh = blockIdx.y, col = threadIdx.x;
  const float* q = ql + ((long long)bh * 256 + row) * 64;
  const float* k = kl + ((long long)bh * 256 + col) * 64;
  float s = 0.f;
#pragma unroll
  for (int d = 0; d < 64; ++d) s += q[d] * k[d];
  __shared__ float red[256];
  red[col] = s; __syncthreads();
  for (int o = 128; o > 0; o >>= 1) { if (col < o) red[col] = fmaxf(red[col], red[col + o]); __syncthreads(); }
  float m = red[0]; __syncthreads();
  float e = expf(s - m);
  red[col] = e; __syncthreads();
  for (int o = 128; o > 0; o >>= 1) { if (col < o) red[col] += red[col + o]; __syncthreads(); }
  a2[((long long)bh * 256 + row) * 256 + col] = e / red[0];
}

__global__ void zero2_k(float* p) { if (threadIdx.x < 2) p[threadIdx.x] = 0.f; }

// Global max of row-sums (tr=0) / col-sums (tr=1) of |attn2| into scal[tr].
__global__ void absmax_rc(const float* __restrict__ a2, float* __restrict__ scal, int tr) {
  int bh = blockIdx.y, idx = blockIdx.x, t = threadIdx.x;
  float v = tr ? a2[((long long)bh * 256 + t) * 256 + idx]
               : a2[((long long)bh * 256 + idx) * 256 + t];
  __shared__ float red[256];
  red[t] = fabsf(v); __syncthreads();
  for (int o = 128; o > 0; o >>= 1) { if (t < o) red[t] += red[t + o]; __syncthreads(); }
  if (t == 0) atomicMax((int*)(scal + tr), __float_as_int(red[0]));  // values >= 0
}

// z0 = attn2^T / (max_rowsum * max_colsum)
__global__ void pinv_init(const float* __restrict__ a2, const float* __restrict__ scal,
                          float* __restrict__ z) {
  int bh = blockIdx.y, i = blockIdx.x, j = threadIdx.x;
  float denom = scal[0] * scal[1];
  z[((long long)bh * 256 + i) * 256 + j] = a2[((long long)bh * 256 + j) * 256 + i] / denom;
}

// t1 = 7I - xz
__global__ void ew_diag7(const float* __restrict__ xz, float* __restrict__ t1) {
  int bh = blockIdx.y, i = blockIdx.x, j = threadIdx.x;
  long long o = ((long long)bh * 256 + i) * 256 + j;
  t1[o] = ((i == j) ? 7.f : 0.f) - xz[o];
}

// Flash-style W = softmax(q_l @ k^T) @ v over N=16384 keys. One wave per 16-row tile.
__global__ void attn3v_flash(const float* __restrict__ qkv, const float* __restrict__ ql,
                             float* __restrict__ Wt) {
  int bh = blockIdx.y, batch = bh >> 3, head = bh & 7;
  int row0 = blockIdx.x * 16;
  int lane = threadIdx.x;
  const float* qrow = ql + ((long long)bh * 256 + row0) * 64;
  v16h aq0 = g_a(qrow, 64, lane, 1.f);
  v16h aq1 = g_a(qrow + 32, 64, lane, 1.f);
  __shared__ float Sc[16][33];
  __shared__ f16 Ps[16][32];
  __shared__ float mrow[16], lrow[16], srow[16];
  if (lane < 16) { mrow[lane] = -1e30f; lrow[lane] = 0.f; }
  __syncthreads();
  v8f O0 = {}, O1 = {}, O2 = {}, O3 = {};
  const float* kbase = qkv + (long long)batch * 16384 * 1536 + 512 + head * 64;
  const float* vbase = qkv + (long long)batch * 16384 * 1536 + 1024 + head * 64;
  const int rb = (lane >> 4) << 3;
  const int cc = lane & 15;
#pragma unroll 1
  for (int kb = 0; kb < 16384; kb += 32) {
    // prefetch next chunk's k/v rows into L2 (global_prefetch_b8)
    if (kb + 32 < 16384) {
      __builtin_prefetch(kbase + (long long)(kb + 32 + lane) * 1536, 0, 1);
      __builtin_prefetch(vbase + (long long)(kb + 32 + lane) * 1536, 0, 1);
    }
#pragma unroll
    for (int nt = 0; nt < 2; ++nt) {
      v8f s = {};
      const float* kp = kbase + (long long)(kb + nt * 16) * 1536;
      s = wmma_f16(aq0, g_bT(kp, 1536, lane), s);
      s = wmma_f16(aq1, g_bT(kp + 32, 1536, lane), s);
      int c = nt * 16 + cc;
#pragma unroll
      for (int r = 0; r < 8; ++r) Sc[rb + r][c] = s[r];
    }
    __syncthreads();
    if (lane < 16) {
      float tm = mrow[lane];
      for (int c = 0; c < 32; ++c) tm = fmaxf(tm, Sc[lane][c]);
      float sc = expf(mrow[lane] - tm);
      float ls = 0.f;
      for (int c = 0; c < 32; ++c) {
        float pv = expf(Sc[lane][c] - tm);
        Ps[lane][c] = (f16)pv;
        ls += pv;
      }
      lrow[lane] = lrow[lane] * sc + ls;
      mrow[lane] = tm;
      srow[lane] = sc;
    }
    __syncthreads();
#pragma unroll
    for (int r = 0; r < 8; ++r) {
      float f = srow[rb + r];
      O0[r] *= f; O1[r] *= f; O2[r] *= f; O3[r] *= f;
    }
    v16h aP = lds_a(&Ps[0][0], 32, lane);
    const float* vp = vbase + (long long)kb * 1536;
    O0 = wmma_f16(aP, g_b(vp + 0, 1536, lane), O0);
    O1 = wmma_f16(aP, g_b(vp + 16, 1536, lane), O1);
    O2 = wmma_f16(aP, g_b(vp + 32, 1536, lane), O2);
    O3 = wmma_f16(aP, g_b(vp + 48, 1536, lane), O3);
    __syncthreads();
  }
#pragma unroll
  for (int r = 0; r < 8; ++r) {
    float inv = 1.f / lrow[rb + r];
    long long o = ((long long)bh * 256 + row0 + rb + r) * 64 + cc;
    Wt[o]      = O0[r] * inv;
    Wt[o + 16] = O1[r] * inv;
    Wt[o + 32] = O2[r] * inv;
    Wt[o + 48] = O3[r] * inv;
  }
}

// Fused out = softmax(q @ k_l^T) @ S  (S = pinv @ (attn3@v), [256,64] per head).
// One wave per 16-token tile; attn1 never materialized.
__global__ void attn1_out(const float* __restrict__ qkv, const float* __restrict__ kl,
                          const float* __restrict__ Smat, float* __restrict__ outa) {
  int bh = blockIdx.y, batch = bh >> 3, head = bh & 7;
  int row0 = blockIdx.x * 16;
  int lane = threadIdx.x;
  const float* qp = qkv + ((long long)batch * 16384 + row0) * 1536 + head * 64;
  v16h aq0 = g_a(qp, 1536, lane, 0.125f);
  v16h aq1 = g_a(qp + 32, 1536, lane, 0.125f);
  __shared__ float Sc[16][257];
  __shared__ f16 Ps[16][256];
  const float* klb = kl + (long long)bh * 256 * 64;
  const int rb = (lane >> 4) << 3;
  const int cc = lane & 15;
#pragma unroll 1
  for (int ct = 0; ct < 16; ++ct) {
    v8f s = {};
    const float* kp = klb + ct * 16 * 64;
    s = wmma_f16(aq0, g_bT(kp, 64, lane), s);
    s = wmma_f16(aq1, g_bT(kp + 32, 64, lane), s);
    int c = ct * 16 + cc;
#pragma unroll
    for (int r = 0; r < 8; ++r) Sc[rb + r][c] = s[r];
  }
  __syncthreads();
  if (lane < 16) {
    float m = -1e30f;
    for (int c = 0; c < 256; ++c) m = fmaxf(m, Sc[lane][c]);
    float ls = 0.f;
    for (int c = 0; c < 256; ++c) { float pv = expf(Sc[lane][c] - m); Sc[lane][c] = pv; ls += pv; }
    float inv = 1.f / ls;
    for (int c = 0; c < 256; ++c) Ps[lane][c] = (f16)(Sc[lane][c] * inv);
  }
  __syncthreads();
  const float* sb = Smat + (long long)bh * 256 * 64;
  v8f O0 = {}, O1 = {}, O2 = {}, O3 = {};
#pragma unroll 1
  for (int ks = 0; ks < 8; ++ks) {
    v16h aP = lds_a(&Ps[0][ks * 32], 256, lane);
    const float* sp = sb + (long long)ks * 32 * 64;
    O0 = wmma_f16(aP, g_b(sp + 0, 64, lane), O0);
    O1 = wmma_f16(aP, g_b(sp + 16, 64, lane), O1);
    O2 = wmma_f16(aP, g_b(sp + 32, 64, lane), O2);
    O3 = wmma_f16(aP, g_b(sp + 48, 64, lane), O3);
  }
#pragma unroll
  for (int r = 0; r < 8; ++r) {
    long long o = ((long long)batch * 16384 + row0 + rb + r) * 512 + head * 64 + cc;
    outa[o]      = O0[r];
    outa[o + 16] = O1[r];
    outa[o + 32] = O2[r];
    outa[o + 48] = O3[r];
  }
}

// Depthwise conv residual over sequence (kernel 33, per-head weight), adds into outa.
__global__ void conv_res(const float* __restrict__ qkv, const float* __restrict__ w,
                         float* __restrict__ outa) {
  long long idx = (long long)blockIdx.x * 256 + threadIdx.x;  // over 2*16384*512
  int c = (int)(idx & 511);
  long long bn = idx >> 9;
  int n = (int)(bn & 16383);
  int batch = (int)(bn >> 14);
  int head = c >> 6;
  const float* wr = w + head * 33;
  const float* vb = qkv + ((long long)batch * 16384) * 1536 + 1024 + c;
  float acc = 0.f;
#pragma unroll 1
  for (int kk = 0; kk < 33; ++kk) {
    int tpos = n + kk - 16;
    if (tpos >= 0 && tpos < 16384) acc += vb[(long long)tpos * 1536] * wr[kk];
  }
  outa[idx] += acc;
}

// MIL pooling chain
__global__ void att2_score(const float* __restrict__ t, const float* __restrict__ w,
                           const float* __restrict__ b, float* __restrict__ sc) {
  int i = blockIdx.x * 256 + threadIdx.x;
  const float* tr = t + (long long)i * 128;
  float s = b[0];
#pragma unroll 1
  for (int d = 0; d < 128; ++d) s += tr[d] * w[d];
  sc[i] = s;
}

__global__ void softmax_n(const float* __restrict__ sc, float* __restrict__ an,
                          float* __restrict__ dout_a) {
  int b = blockIdx.x, t = threadIdx.x;
  const float* s = sc + b * 16384;
  float m = -1e30f;
  for (int i = t; i < 16384; i += 1024) m = fmaxf(m, s[i]);
  __shared__ float red[1024];
  red[t] = m; __syncthreads();
  for (int o = 512; o > 0; o >>= 1) { if (t < o) red[t] = fmaxf(red[t], red[t + o]); __syncthreads(); }
  m = red[0]; __syncthreads();
  float ls = 0.f;
  for (int i = t; i < 16384; i += 1024) ls += expf(s[i] - m);
  red[t] = ls; __syncthreads();
  for (int o = 512; o > 0; o >>= 1) { if (t < o) red[t] += red[t + o]; __syncthreads(); }
  float inv = 1.f / red[0];
  for (int i = t; i < 16384; i += 1024) {
    float v = expf(s[i] - m) * inv;
    an[b * 16384 + i] = v;
    dout_a[b * 16384 + i] = v;
  }
}

__global__ void pool_z(const float* __restrict__ an, const float* __restrict__ h,
                       float* __restrict__ zp) {
  int c = blockIdx.x * 256 + threadIdx.x;  // 0..1023
  int b = c >> 9, ch = c & 511;
  const float* hr = h + (long long)b * 16384 * 512 + ch;
  const float* ar = an + b * 16384;
  float s = 0.f;
  for (int n = 0; n < 16384; ++n) s += ar[n] * hr[(long long)n * 512];
  zp[c] = s;
}

__global__ void head_k(const float* __restrict__ zp, const float* __restrict__ w,
                       const float* __restrict__ bias, float* __restrict__ dout) {
  int t = threadIdx.x;
  __shared__ float lg[4];
  if (t < 4) {
    int b = t >> 1, c = t & 1;
    float s = bias[c];
    for (int d = 0; d < 512; ++d) s += zp[b * 512 + d] * w[d * 2 + c];
    lg[t] = s;
    dout[t] = s;                 // logits [2,2]
  }
  __syncthreads();
  if (t < 2) {
    float a = lg[t * 2], b2 = lg[t * 2 + 1];
    float m = fmaxf(a, b2);
    float ea = expf(a - m), eb = expf(b2 - m), s = ea + eb;
    dout[4 + t * 2] = ea / s;    // Y_prob [2,2]
    dout[4 + t * 2 + 1] = eb / s;
  }
}

// ---------------- Host orchestration ----------------
extern "C" void kernel_launch(void* const* d_in, const int* in_sizes, int n_in,
                              void* d_out, int out_size, void* d_ws, size_t ws_size,
                              hipStream_t stream) {
  (void)in_sizes; (void)n_in; (void)out_size; (void)ws_size;
  const float* X      = (const float*)d_in[0];
  const float* fc1_w  = (const float*)d_in[1];
  const float* fc1_b  = (const float*)d_in[2];
  const float* ln1_g  = (const float*)d_in[3];
  const float* ln1_b  = (const float*)d_in[4];
  const float* qkv_w  = (const float*)d_in[5];
  const float* out_w  = (const float*)d_in[6];
  const float* out_b  = (const float*)d_in[7];
  const float* conv_w = (const float*)d_in[8];
  const float* ln2_g  = (const float*)d_in[9];
  const float* ln2_b  = (const float*)d_in[10];
  const float* ff1_w  = (const float*)d_in[11];
  const float* ff1_b  = (const float*)d_in[12];
  const float* ff2_w  = (const float*)d_in[13];
  const float* ff2_b  = (const float*)d_in[14];
  const float* att1_w = (const float*)d_in[15];
  const float* att1_b = (const float*)d_in[16];
  const float* att2_w = (const float*)d_in[17];
  const float* att2_b = (const float*)d_in[18];
  const float* fc2_w  = (const float*)d_in[19];
  const float* fc2_b  = (const float*)d_in[20];
  float* out = (float*)d_out;

  float* p = (float*)d_ws;
  auto take = [&](size_t n) { float* r = p; p += n; return r; };
  float* H    = take(32768ull * 512);   // residual stream
  float* SH   = take(32768ull * 512);   // layernorm output
  float* OA   = take(32768ull * 512);   // attention output (pre out-proj)
  float* BIGB = take(32768ull * 1536);  // qkv / FFN-mid chunks / pooling features
  float* QL   = take(16ull * 256 * 64);
  float* KL   = take(16ull * 256 * 64);
  float* A2   = take(16ull * 256 * 256);
  float* Z    = take(16ull * 256 * 256);
  float* Z2   = take(16ull * 256 * 256);
  float* XZ   = take(16ull * 256 * 256);
  float* T1   = take(16ull * 256 * 256);
  float* T2   = take(16ull * 256 * 256);
  float* T3   = take(16ull * 256 * 256);
  float* WT   = take(16ull * 256 * 64);
  float* SM   = take(16ull * 256 * 64);
  float* SCR  = take(32768);
  float* AN   = take(32768);
  float* ZP   = take(1024);
  float* SCAL = take(2);

  auto gemm = [&](const float* A, const float* B, const float* bias, const float* resid,
                  float* C, int M, int N, int K, long long sA, long long sB, long long sC,
                  int batch, float scale, float diag, int act) {
    dim3 g((N + 127) / 128, (M + 127) / 128, batch);
    if ((M % 128 == 0) && (N % 128 == 0))
      gemm_wmma<true><<<g, 256, 0, stream>>>(A, B, bias, resid, C, M, N, K, sA, sB, sC,
                                             scale, diag, act);
    else
      gemm_wmma<false><<<g, 256, 0, stream>>>(A, B, bias, resid, C, M, N, K, sA, sB, sC,
                                              scale, diag, act);
  };

  // h = relu(x @ fc1_w + fc1_b)
  gemm(X, fc1_w, fc1_b, nullptr, H, 32768, 512, 1024, 0, 0, 0, 1, 1.f, 0.f, 1);

  for (int i = 0; i < 2; ++i) {
    layernorm512<<<32768, 256, 0, stream>>>(H, ln1_g + i * 512, ln1_b + i * 512, SH);
    // qkv = ln(h) @ qkv_w[i]
    gemm(SH, qkv_w + (size_t)i * 512 * 1536, nullptr, nullptr, BIGB,
         32768, 1536, 512, 0, 0, 0, 1, 1.f, 0.f, 0);
    landmarks_k<<<4096, 64, 0, stream>>>(BIGB, QL, KL);
    attn2_softmax<<<dim3(256, 16), 256, 0, stream>>>(QL, KL, A2);
    // Newton-Schulz pseudo-inverse of attn2
    zero2_k<<<1, 32, 0, stream>>>(SCAL);
    absmax_rc<<<dim3(256, 16), 256, 0, stream>>>(A2, SCAL, 0);
    absmax_rc<<<dim3(256, 16), 256, 0, stream>>>(A2, SCAL, 1);
    pinv_init<<<dim3(256, 16), 256, 0, stream>>>(A2, SCAL, Z);
    float* za = Z; float* zb = Z2;
    for (int it = 0; it < 6; ++it) {
      gemm(A2, za, nullptr, nullptr, XZ, 256, 256, 256, 65536, 65536, 65536, 16, 1.f, 0.f, 0);
      ew_diag7<<<dim3(256, 16), 256, 0, stream>>>(XZ, T1);
      gemm(XZ, T1, nullptr, nullptr, T2, 256, 256, 256, 65536, 65536, 65536, 16, -1.f, 15.f, 0);
      gemm(XZ, T2, nullptr, nullptr, T3, 256, 256, 256, 65536, 65536, 65536, 16, -1.f, 13.f, 0);
      gemm(za, T3, nullptr, nullptr, zb, 256, 256, 256, 65536, 65536, 65536, 16, 0.25f, 0.f, 0);
      float* tmp = za; za = zb; zb = tmp;
    }
    // W = softmax(q_l @ k^T) @ v  (streaming, never materializes attn3)
    attn3v_flash<<<dim3(16, 16), 32, 0, stream>>>(BIGB, QL, WT);
    // S = pinv @ W
    gemm(za, WT, nullptr, nullptr, SM, 256, 64, 256, 65536, 16384, 16384, 16, 1.f, 0.f, 0);
    // out = softmax(q @ k_l^T) @ S   (never materializes attn1)
    attn1_out<<<dim3(1024, 16), 32, 0, stream>>>(BIGB, KL, SM, OA);
    conv_res<<<65536, 256, 0, stream>>>(BIGB, conv_w + i * 8 * 33, OA);
    // h = h + out @ out_w + out_b
    gemm(OA, out_w + (size_t)i * 512 * 512, out_b + i * 512, H, H,
         32768, 512, 512, 0, 0, 0, 1, 1.f, 0.f, 0);
    // FFN (chunked over rows so the 2048-wide mid reuses the qkv buffer)
    layernorm512<<<32768, 256, 0, stream>>>(H, ln2_g + i * 512, ln2_b + i * 512, SH);
    for (int ch = 0; ch < 4; ++ch) {
      const float* ain = SH + (size_t)ch * 8192 * 512;
      float* hch = H + (size_t)ch * 8192 * 512;
      gemm(ain, ff1_w + (size_t)i * 512 * 2048, ff1_b + i * 2048, nullptr, BIGB,
           8192, 2048, 512, 0, 0, 0, 1, 1.f, 0.f, 2);
      gemm(BIGB, ff2_w + (size_t)i * 2048 * 512, ff2_b + i * 512, hch, hch,
           8192, 512, 2048, 0, 0, 0, 1, 1.f, 0.f, 0);
    }
  }

  // Gated-attention MIL pooling + classifier head
  gemm(H, att1_w, att1_b, nullptr, BIGB, 32768, 128, 512, 0, 0, 0, 1, 1.f, 0.f, 3);
  att2_score<<<128, 256, 0, stream>>>(BIGB, att2_w, att2_b, SCR);
  softmax_n<<<2, 1024, 0, stream>>>(SCR, AN, out + 8);   // a -> d_out[8..]
  pool_z<<<4, 256, 0, stream>>>(AN, H, ZP);
  head_k<<<1, 32, 0, stream>>>(ZP, fc2_w, fc2_b, out);   // logits + Y_prob -> d_out[0..7]
}
